// VerifyPRM_31250182046379
// MI455X (gfx1250) — matrix operand order repaired and checked by reading, not compiled, not run on hardware
//
#include <hip/hip_runtime.h>

// ---------------------------------------------------------------------------
// Types for CDNA5 WMMA (gfx1250, wave32)
// ---------------------------------------------------------------------------
typedef __attribute__((ext_vector_type(16))) __bf16          v16bf;
typedef __attribute__((ext_vector_type(8)))  float           v8f;
typedef __attribute__((ext_vector_type(8)))  unsigned short  us8;

union Frag16 {
    v16bf v;
    us8   h[2];
};

// Pack two fp32 -> one dword holding {bf16(y)<<16 | bf16(x)}, RNE rounding,
// packed with a single v_perm_b32.
__device__ __forceinline__ unsigned int pack2_bf16(float x, float y) {
    unsigned int ux = __float_as_uint(x);
    unsigned int uy = __float_as_uint(y);
    ux += 0x7FFFu + ((ux >> 16) & 1u);
    uy += 0x7FFFu + ((uy >> 16) & 1u);
    // sel 0x07060302: b0=src1.b2 b1=src1.b3 b2=src0.b2 b3=src0.b3
    return __builtin_amdgcn_perm(uy, ux, 0x07060302u);
}

// wave32 cross-lane sum via ds_swizzle (group-of-32 xor mode)
template <int IMM>
__device__ __forceinline__ float swz_add(float x) {
    return x + __int_as_float(__builtin_amdgcn_ds_swizzle(__float_as_int(x), IMM));
}
__device__ __forceinline__ float wave_reduce_add(float x) {
    x = swz_add<0x041F>(x);  // xor 1
    x = swz_add<0x081F>(x);  // xor 2
    x = swz_add<0x101F>(x);  // xor 4
    x = swz_add<0x201F>(x);  // xor 8
    x = swz_add<0x401F>(x);  // xor 16
    return x;
}

// ---------------------------------------------------------------------------
// Generic NT GEMM: C[M,N] (=|+=) alpha * A[M,K] * Bw[N,K]^T
// fp32 in memory, bf16 WMMA compute with f32 accumulate.
// Block tile 128x64, K-step 32, 8 waves (4x2), each wave 32x32 (4 wmma accs).
// Software-pipelined: next K-tile prefetched into registers while current
// tile computes; LDS ping-pong buffers -> one barrier per K-step.
// Grid is (M-blocks, N-blocks): x-major dispatch makes the 16 consecutive
// blocks share one 64-row B (weight) panel -> weights stream from HBM once,
// A (32 MB activations) is L2-resident and absorbs the re-reads.
// All call sites have M%128==0, N%64==0, K%32==0 -> no bounds checks,
// EXEC stays all-ones (WMMA requirement).
// ---------------------------------------------------------------------------
#define TM 128
#define TN 64
#define TK 32

__global__ __launch_bounds__(256) void gemm_nt_wmma(
    const float* __restrict__ A, const float* __restrict__ Bw,
    float* __restrict__ C, int M_, int N_, int K_, float alpha, int accum)
{
    (void)M_;
    __shared__ alignas(16) unsigned short lA[2][TM * TK];  // 2 x 8 KB bf16
    __shared__ alignas(16) unsigned short lB[2][TN * TK];  // 2 x 4 KB bf16

    const int tid  = threadIdx.x;
    const int lane = tid & 31;
    const int wave = tid >> 5;     // 0..7
    const int wm   = wave & 3;     // 4 waves along M
    const int wn   = wave >> 2;    // 2 waves along N
    const int m0   = blockIdx.x * TM;   // M fastest -> B-panel reuse in L2
    const int n0   = blockIdx.y * TN;
    const int lhi  = lane >> 4;    // 0/1
    const int llo  = lane & 15;

    // Per-thread staging coordinates (constant across K-steps)
    const float* pAg[4];
    int sAoff[4];
    #pragma unroll
    for (int r = 0; r < 4; ++r) {
        int idx = tid + r * 256;           // float4 index (1024 total)
        int row = idx >> 3;                // TK/4 = 8 float4 per row
        int c4  = (idx & 7) << 2;
        pAg[r]   = A + (size_t)(m0 + row) * K_ + c4;
        sAoff[r] = row * TK + c4;
    }
    const float* pBg[2];
    int sBoff[2];
    #pragma unroll
    for (int r = 0; r < 2; ++r) {
        int idx = tid + r * 256;           // 512 total
        int row = idx >> 3;
        int c4  = (idx & 7) << 2;
        pBg[r]   = Bw + (size_t)(n0 + row) * K_ + c4;
        sBoff[r] = row * TK + c4;
    }

    v8f acc[2][2];
    acc[0][0] = {}; acc[0][1] = {}; acc[1][0] = {}; acc[1][1] = {};

    // Prologue: fetch K-tile 0 into registers
    float4 ra[4], rb[2];
    #pragma unroll
    for (int r = 0; r < 4; ++r) ra[r] = *(const float4*)(pAg[r]);
    #pragma unroll
    for (int r = 0; r < 2; ++r) rb[r] = *(const float4*)(pBg[r]);

    const int nk = K_ / TK;
    for (int t = 0; t < nk; ++t) {
        unsigned short* sA = lA[t & 1];
        unsigned short* sB = lB[t & 1];

        // Convert staged registers fp32 -> bf16 (v_perm pack), store to LDS
        #pragma unroll
        for (int r = 0; r < 4; ++r) {
            unsigned int* d = (unsigned int*)&sA[sAoff[r]];
            d[0] = pack2_bf16(ra[r].x, ra[r].y);
            d[1] = pack2_bf16(ra[r].z, ra[r].w);
        }
        #pragma unroll
        for (int r = 0; r < 2; ++r) {
            unsigned int* d = (unsigned int*)&sB[sBoff[r]];
            d[0] = pack2_bf16(rb[r].x, rb[r].y);
            d[1] = pack2_bf16(rb[r].z, rb[r].w);
        }
        __syncthreads();

        // Prefetch next K-tile into registers (latency hidden under compute)
        if (t + 1 < nk) {
            int ko = (t + 1) * TK;
            #pragma unroll
            for (int r = 0; r < 4; ++r) ra[r] = *(const float4*)(pAg[r] + ko);
            #pragma unroll
            for (int r = 0; r < 2; ++r) rb[r] = *(const float4*)(pBg[r] + ko);
            if (t + 2 < nk) {  // L2 prefetch at distance 2 -> global_prefetch_b8
                __builtin_prefetch(pAg[0] + (t + 2) * TK, 0, 1);
                __builtin_prefetch(pBg[0] + (t + 2) * TK, 0, 1);
            }
        }

        // Build fragments per ISA 7.12.2 VGPR layouts.
        // A (16x32, M x K): lane l holds row M=l%16;
        //   half j -> k = (j>=8 ? 16 : 0) + (l>=16 ? 8 : 0) + (j&7)
        Frag16 a[2], b[2];
        #pragma unroll
        for (int i = 0; i < 2; ++i) {
            const us8* p = (const us8*)&sA[(wm * 32 + i * 16 + llo) * TK + lhi * 8];
            a[i].h[0] = p[0];   // k = lhi*8 + 0..7
            a[i].h[1] = p[2];   // k = 16 + lhi*8 + 0..7
        }
        // B (32x16, K x N): lane l holds column N=l%16; half j -> k = (j>=8 ? 16 : 0) + (l>=16 ? 8 : 0) + (j&7)
        #pragma unroll
        for (int j = 0; j < 2; ++j) {
            const us8* p = (const us8*)&sB[(wn * 32 + j * 16 + llo) * TK + lhi * 8];
            b[j].h[0] = p[0];
            b[j].h[1] = p[2];
        }
        #pragma unroll
        for (int i = 0; i < 2; ++i)
            #pragma unroll
            for (int j = 0; j < 2; ++j)
                acc[i][j] = __builtin_amdgcn_wmma_f32_16x16x32_bf16(
                    false, a[i].v, false, b[j].v, (short)0, acc[i][j], false, false);
    }

    // C/D layout: VGPR r -> row = r + 8*(lane>=16), col = lane%16
    #pragma unroll
    for (int i = 0; i < 2; ++i)
        #pragma unroll
        for (int j = 0; j < 2; ++j)
            #pragma unroll
            for (int r = 0; r < 8; ++r) {
                int row = m0 + wm * 32 + i * 16 + lhi * 8 + r;
                int col = n0 + wn * 32 + j * 16 + llo;
                size_t off = (size_t)row * N_ + col;
                float vv = alpha * acc[i][j][r];
                if (accum) C[off] += vv; else C[off] = vv;
            }
}

// ---------------------------------------------------------------------------
// RMSNorm over D=4096 per row. One 256-thread block per row.
// ---------------------------------------------------------------------------
__global__ __launch_bounds__(256) void rmsnorm_k(
    const float* __restrict__ x, const float* __restrict__ w,
    float* __restrict__ out, long in_stride, long out_stride)
{
    const float* xr = x + (size_t)blockIdx.x * in_stride;
    float ss = 0.f;
    for (int i = threadIdx.x; i < 4096; i += 256) { float v = xr[i]; ss += v * v; }
    __shared__ float red[256];
    red[threadIdx.x] = ss;
    __syncthreads();
    for (int s = 128; s > 0; s >>= 1) {
        if (threadIdx.x < s) red[threadIdx.x] += red[threadIdx.x + s];
        __syncthreads();
    }
    float inv = rsqrtf(red[0] * (1.0f / 4096.0f) + 1e-5f);
    float* orow = out + (size_t)blockIdx.x * out_stride;
    for (int i = threadIdx.x; i < 4096; i += 256) orow[i] = xr[i] * inv * w[i];
}

// ---------------------------------------------------------------------------
// RoPE in-place; one thread per (b,s,h,i2) pair, i2 in [0,64)
// ---------------------------------------------------------------------------
__global__ __launch_bounds__(256) void rope_k(float* __restrict__ q)
{
    size_t i = (size_t)blockIdx.x * 256 + threadIdx.x;   // B*S*H*64 pairs
    int i2   = (int)(i & 63);
    int h    = (int)((i >> 6) & 31);
    int srow = (int)(i >> 11);          // b*S + s
    int pos  = srow & 1023;             // S = 1024
    size_t base = (size_t)srow * 4096 + (size_t)h * 128 + i2;
    float inv = __expf(-9.2103403719761836f * ((float)i2 * (1.0f / 64.0f))); // 10000^(-i2/64)
    float ang = (float)pos * inv;
    float c = __cosf(ang), s = __sinf(ang);
    float x1 = q[base], x2 = q[base + 64];
    q[base]      = x1 * c - x2 * s;
    q[base + 64] = x2 * c + x1 * s;
}

// ---------------------------------------------------------------------------
// Streaming causal attention (online softmax). One wave per (b,h,q) row;
// each lane owns 4 of HD=128 channels. K/V stream through L2 (64 MB/layer).
// ---------------------------------------------------------------------------
__global__ __launch_bounds__(256) void attn_k(
    const float* __restrict__ q, const float* __restrict__ k,
    const float* __restrict__ v, const int* __restrict__ amask,
    float* __restrict__ o)
{
    int gwave = (int)(((size_t)blockIdx.x * 256 + threadIdx.x) >> 5);
    int lane  = threadIdx.x & 31;
    int qpos  = gwave & 1023;
    int bh    = gwave >> 10;
    int h     = bh & 31;
    int b     = bh >> 5;
    size_t rowq = ((size_t)(b * 1024 + qpos)) * 4096 + (size_t)h * 128 + lane * 4;
    float4 qv = *(const float4*)(q + rowq);
    float m = -3.0e38f, l = 0.f;
    float4 acc = {0.f, 0.f, 0.f, 0.f};
    const float scale = 0.08838834764831845f;  // 1/sqrt(128)
    for (int j = 0; j <= qpos; ++j) {
        size_t rk = ((size_t)(b * 1024 + j)) * 4096 + (size_t)h * 128 + lane * 4;
        float4 kv = *(const float4*)(k + rk);
        float p = qv.x * kv.x + qv.y * kv.y + qv.z * kv.z + qv.w * kv.w;
        p = wave_reduce_add(p);
        float sc = p * scale + (amask[b * 1024 + j] > 0 ? 0.f : -1.0e9f);
        float mn   = fmaxf(m, sc);
        float corr = __expf(m - mn);
        float pe   = __expf(sc - mn);
        float4 vv = *(const float4*)(v + rk);
        acc.x = acc.x * corr + pe * vv.x;
        acc.y = acc.y * corr + pe * vv.y;
        acc.z = acc.z * corr + pe * vv.z;
        acc.w = acc.w * corr + pe * vv.w;
        l = l * corr + pe;
        m = mn;
    }
    float invl = 1.0f / l;
    float4 ov = {acc.x * invl, acc.y * invl, acc.z * invl, acc.w * invl};
    *(float4*)(o + rowq) = ov;
}

// ---------------------------------------------------------------------------
// g = silu(g) * u, elementwise
// ---------------------------------------------------------------------------
__global__ __launch_bounds__(256) void silu_mul_k(float* __restrict__ g,
                                                  const float* __restrict__ u)
{
    size_t i = (size_t)blockIdx.x * 256 + threadIdx.x;
    float x = g[i];
    float s = x / (1.0f + __expf(-x));
    g[i] = s * u[i];
}

// ---------------------------------------------------------------------------
// Embedding gather: x[row, :] = embed[ids[row], :]
// ---------------------------------------------------------------------------
__global__ __launch_bounds__(256) void embed_k(const int* __restrict__ ids,
    const float* __restrict__ emb, float* __restrict__ x)
{
    size_t i = (size_t)blockIdx.x * 256 + threadIdx.x;  // over 2048*4096
    int row = (int)(i >> 12);
    int col = (int)(i & 4095);
    x[i] = emb[(size_t)ids[row] * 4096 + col];
}

// ---------------------------------------------------------------------------
// Classifier head: out[b,c] = dot(xn[b,:], W[c,:]) + bias[c]; 6 waves.
// ---------------------------------------------------------------------------
__global__ void cls_k(const float* __restrict__ xn, const float* __restrict__ W,
                      const float* __restrict__ bias, float* __restrict__ out)
{
    int wave = threadIdx.x >> 5, lane = threadIdx.x & 31;
    int b = wave / 3, c = wave % 3;
    float s = 0.f;
    for (int i = lane; i < 4096; i += 32) s += xn[b * 4096 + i] * W[c * 4096 + i];
    s = wave_reduce_add(s);
    if (wave < 6 && lane == 0) out[b * 3 + c] = s + bias[c];
}

// ---------------------------------------------------------------------------
// Host orchestration
// ---------------------------------------------------------------------------
extern "C" void kernel_launch(void* const* d_in, const int* in_sizes, int n_in,
                              void* d_out, int out_size, void* d_ws, size_t ws_size,
                              hipStream_t stream)
{
    (void)in_sizes; (void)n_in; (void)out_size; (void)ws_size;
    const int*   input_ids = (const int*)d_in[0];
    const int*   amask     = (const int*)d_in[1];
    const float* embed     = (const float*)d_in[2];
    const float* Wq = (const float*)d_in[3];
    const float* Wk = (const float*)d_in[4];
    const float* Wv = (const float*)d_in[5];
    const float* Wo = (const float*)d_in[6];
    const float* Aq = (const float*)d_in[7];
    const float* Bq = (const float*)d_in[8];
    const float* Ak = (const float*)d_in[9];
    const float* Bk = (const float*)d_in[10];
    const float* Av = (const float*)d_in[11];
    const float* Bv = (const float*)d_in[12];
    const float* Ao = (const float*)d_in[13];
    const float* Bo = (const float*)d_in[14];
    const float* Wgate = (const float*)d_in[15];
    const float* Wup   = (const float*)d_in[16];
    const float* Wdown = (const float*)d_in[17];
    const float* attn_nw  = (const float*)d_in[18];
    const float* mlp_nw   = (const float*)d_in[19];
    const float* final_nw = (const float*)d_in[20];
    const float* cls_W    = (const float*)d_in[21];
    const float* cls_b    = (const float*)d_in[22];
    float* out = (float*)d_out;

    const int Dm = 4096, Fm = 11008, Rm = 64, Mr = 2048;
    float* ws = (float*)d_ws;
    float* x  = ws;                       // [2048,4096] residual stream
    float* h  = x  + (size_t)Mr * Dm;     // [2048,4096] normed / attn-out scratch
    float* qb = h  + (size_t)Mr * Dm;     // [2048,4096]
    float* kb = qb + (size_t)Mr * Dm;     // [2048,4096]
    float* vb = kb + (size_t)Mr * Dm;     // [2048,4096]
    float* t  = vb + (size_t)Mr * Dm;     // [2048,64]  LoRA temp
    float* g  = t  + (size_t)Mr * Rm;     // [2048,11008]
    float* u  = g  + (size_t)Mr * Fm;     // [2048,11008]

    const float lora_s = 16.0f / 64.0f;   // ALPHA / R

    auto gemm = [&](const float* Am, const float* Bm, float* Cm,
                    int N_, int K_, float alpha, int accum) {
        dim3 grid(Mr / TM, N_ / TN);      // M fastest: B-panel L2 reuse
        gemm_nt_wmma<<<grid, 256, 0, stream>>>(Am, Bm, Cm, Mr, N_, K_, alpha, accum);
    };

    embed_k<<<(Mr * Dm) / 256, 256, 0, stream>>>(input_ids, embed, x);

    for (int l = 0; l < 2; ++l) {
        size_t wo = (size_t)l * Dm * Dm;
        size_t ao = (size_t)l * Rm * Dm;
        size_t bo = (size_t)l * Dm * Rm;
        size_t fo = (size_t)l * (size_t)Fm * Dm;

        rmsnorm_k<<<Mr, 256, 0, stream>>>(x, attn_nw + (size_t)l * Dm, h, Dm, Dm);

        // q/k/v = h @ W^T + (alpha/R) * (h @ A^T) @ B^T
        gemm(h, Aq + ao, t,  Rm, Dm, 1.0f, 0);
        gemm(h, Wq + wo, qb, Dm, Dm, 1.0f, 0);
        gemm(t, Bq + bo, qb, Dm, Rm, lora_s, 1);

        gemm(h, Ak + ao, t,  Rm, Dm, 1.0f, 0);
        gemm(h, Wk + wo, kb, Dm, Dm, 1.0f, 0);
        gemm(t, Bk + bo, kb, Dm, Rm, lora_s, 1);

        gemm(h, Av + ao, t,  Rm, Dm, 1.0f, 0);
        gemm(h, Wv + wo, vb, Dm, Dm, 1.0f, 0);
        gemm(t, Bv + bo, vb, Dm, Rm, lora_s, 1);

        rope_k<<<16384, 256, 0, stream>>>(qb);
        rope_k<<<16384, 256, 0, stream>>>(kb);

        attn_k<<<8192, 256, 0, stream>>>(qb, kb, vb, amask, h);   // o -> h

        // x += o @ Wo^T + (alpha/R) * (o @ Ao^T) @ Bo^T
        gemm(h, Ao + ao, t, Rm, Dm, 1.0f, 0);
        gemm(h, Wo + wo, x, Dm, Dm, 1.0f, 1);
        gemm(t, Bo + bo, x, Dm, Rm, lora_s, 1);

        // MLP: x += (silu(h@Wg^T) * (h@Wu^T)) @ Wd^T
        rmsnorm_k<<<Mr, 256, 0, stream>>>(x, mlp_nw + (size_t)l * Dm, h, Dm, Dm);
        gemm(h, Wgate + fo, g, Fm, Dm, 1.0f, 0);
        gemm(h, Wup   + fo, u, Fm, Dm, 1.0f, 0);
        silu_mul_k<<<(Mr * Fm) / 256, 256, 0, stream>>>(g, u);
        gemm(g, Wdown + fo, x, Dm, Fm, 1.0f, 1);
    }

    // Final norm only on the two s==0 rows, then 2x3 classifier.
    rmsnorm_k<<<2, 256, 0, stream>>>(x, final_nw, h, (long)1024 * Dm, Dm);
    cls_k<<<1, 192, 0, stream>>>(h, cls_W, cls_b, out);
}
